// GPTQLinear_81381040325023
// MI455X (gfx1250) — matrix operand order repaired and checked by reading, not compiled, not run on hardware
//
#include <hip/hip_runtime.h>

// ---- CDNA5 vector types for WMMA ----
typedef _Float16 v8h  __attribute__((ext_vector_type(8)));
typedef _Float16 v16h __attribute__((ext_vector_type(16)));
typedef float    v8f  __attribute__((ext_vector_type(8)));

#define BM 128
#define BN 128
#define BK 128            // == GROUP_SIZE, so one scale group per K-tile
#define LDT 136           // padded LDS row stride in f16 (272B -> rotates banks)
#define TILE_ELEMS (BM * LDT)
#define THREADS 256

// ------------------------------------------------------------------
// Pre-pass 1: x fp32 -> f16 (exactly the dtype the matrix pipe wants)
// ------------------------------------------------------------------
__global__ __launch_bounds__(256) void cvt_x_f16_kernel(
    const float* __restrict__ x, _Float16* __restrict__ xh, size_t n8) {
  size_t i = (size_t)blockIdx.x * 256 + threadIdx.x;
  if (i >= n8) return;
  size_t base = i * 8;
  const float4* p = (const float4*)(x + base);
  float4 f0 = p[0];
  float4 f1 = p[1];
  v8h h;
  h[0] = (_Float16)f0.x; h[1] = (_Float16)f0.y;
  h[2] = (_Float16)f0.z; h[3] = (_Float16)f0.w;
  h[4] = (_Float16)f1.x; h[5] = (_Float16)f1.y;
  h[6] = (_Float16)f1.z; h[7] = (_Float16)f1.w;
  *(v8h*)(xh + base) = h;
}

// ------------------------------------------------------------------
// Pre-pass 2: group-wise dequant int4-in-int32 -> f16
// wh[n][k] = (f16)( qweight[n][k] * scales[n][k/128] )
// ------------------------------------------------------------------
__global__ __launch_bounds__(256) void dequant_w_f16_kernel(
    const int* __restrict__ q, const float* __restrict__ s,
    _Float16* __restrict__ wh, int in_f, int num_groups, size_t n8) {
  size_t i = (size_t)blockIdx.x * 256 + threadIdx.x;
  if (i >= n8) return;
  size_t base = i * 8;                 // 8 consecutive k of the same row/group
  size_t n = base / (size_t)in_f;
  int    k = (int)(base % (size_t)in_f);
  float sc = s[n * (size_t)num_groups + (k >> 7)];
  const int4* p = (const int4*)(q + base);
  int4 q0 = p[0];
  int4 q1 = p[1];
  v8h h;
  h[0] = (_Float16)((float)q0.x * sc); h[1] = (_Float16)((float)q0.y * sc);
  h[2] = (_Float16)((float)q0.z * sc); h[3] = (_Float16)((float)q0.w * sc);
  h[4] = (_Float16)((float)q1.x * sc); h[5] = (_Float16)((float)q1.y * sc);
  h[6] = (_Float16)((float)q1.z * sc); h[7] = (_Float16)((float)q1.w * sc);
  *(v8h*)(wh + base) = h;
}

// Async copy: 16 bytes per lane, global -> LDS, tracked by ASYNCcnt.
// VDST vgpr = LDS byte offset, GVS addressing: saddr base + 32-bit vaddr offset.
__device__ __forceinline__ void async_copy_b128(unsigned lds_off,
                                                const void* base,
                                                unsigned voff) {
  asm volatile("global_load_async_to_lds_b128 %0, %1, %2"
               :: "v"(lds_off), "v"(voff), "s"(base)
               : "memory");
}

__device__ __forceinline__ void wait_async_all() {
  asm volatile("s_wait_asynccnt 0x0" ::: "memory");
}

// ------------------------------------------------------------------
// Main GEMM: out[M][N] = xh[M][K] * wh[N][K]^T + bias
// 128x128 block, 8 waves, each wave 32(M) x 64(N) = 2x4 WMMA tiles.
// Double-buffered LDS fed by async global->LDS DMA (no staging VGPRs).
// ------------------------------------------------------------------
__global__ __launch_bounds__(THREADS) void gptq_wmma_gemm_kernel(
    const _Float16* __restrict__ xh,  // [M][K]
    const _Float16* __restrict__ wh,  // [N][K]
    const float* __restrict__ bias,   // [N]
    float* __restrict__ out,          // [M][N]
    int M, int N, int K) {
  extern __shared__ _Float16 smem[];
  // LDS layout: [buf0 A][buf0 B][buf1 A][buf1 B]

  const int m0 = blockIdx.y * BM;
  const int n0 = blockIdx.x * BN;

  const int t    = threadIdx.x;
  const int lane = t & 31;
  const int wv   = t >> 5;          // wave 0..7
  const int wm   = (wv & 3) * 32;   // 4 waves along M
  const int wn   = (wv >> 2) * 64;  // 2 waves along N
  const int lr   = lane & 15;       // lane row within 16x16 tile
  const int lh   = lane >> 4;       // half-select (ISA f16 A/B layout)

  // staging coords: 256 threads x 16B chunks; 16 chunks per 128-elem row
  const int srow = t >> 4;          // 0..15 (stride 16 over 128 rows)
  const int scol = (t & 15) * 8;    // f16 column of 16B chunk

  // LDS byte offset of this workgroup's shared base (generic ptr low 32 bits)
  const unsigned smem_off = (unsigned)(uintptr_t)(void*)smem;

  v8f acc[2][4] = {};               // fp32 accumulators (WMMA C/D)

  auto issue_async = [&](int buf, int k0) {
    const unsigned abase = smem_off + (unsigned)buf * (2u * TILE_ELEMS * 2u);
    const unsigned bbase = abase + (unsigned)(TILE_ELEMS * 2);
#pragma unroll
    for (int i = 0; i < 8; i++) {
      int r = srow + i * 16;
      unsigned lds_elem = (unsigned)(r * LDT + scol) * 2u;
      unsigned voffA = (unsigned)((m0 + r) * K + k0 + scol) * 2u;
      unsigned voffB = (unsigned)((n0 + r) * K + k0 + scol) * 2u;
      async_copy_b128(abase + lds_elem, xh, voffA);
      async_copy_b128(bbase + lds_elem, wh, voffB);
    }
  };

  auto compute = [&](int buf) {
    const _Float16* A = smem + buf * (2 * TILE_ELEMS);
    const _Float16* B = A + TILE_ELEMS;
#pragma unroll
    for (int kk = 0; kk < BK; kk += 32) {
      v16h a[2], b[4];
#pragma unroll
      for (int tm = 0; tm < 2; tm++) {
        // A 16x32 f16 layout: lanes 0-15 hold K 0-7 & 16-23, lanes 16-31 hold 8-15 & 24-31
        const _Float16* p = A + (wm + tm * 16 + lr) * LDT + kk + lh * 8;
        v8h lo = *(const v8h*)p;
        v8h hi = *(const v8h*)(p + 16);
        a[tm] = __builtin_shufflevector(lo, hi, 0, 1, 2, 3, 4, 5, 6, 7,
                                                8, 9, 10, 11, 12, 13, 14, 15);
      }
#pragma unroll
      for (int tn = 0; tn < 4; tn++) {
        // B 32x16 f16 layout: lanes 0-15 (N=lr) hold K 0-7 & 16-23, lanes 16-31 hold K 8-15 & 24-31
        const _Float16* p = B + (wn + tn * 16 + lr) * LDT + kk + lh * 8;
        v8h lo = *(const v8h*)p;
        v8h hi = *(const v8h*)(p + 16);
        b[tn] = __builtin_shufflevector(lo, hi, 0, 1, 2, 3, 4, 5, 6, 7,
                                                8, 9, 10, 11, 12, 13, 14, 15);
      }
#pragma unroll
      for (int tm = 0; tm < 2; tm++)
#pragma unroll
        for (int tn = 0; tn < 4; tn++)
          acc[tm][tn] = __builtin_amdgcn_wmma_f32_16x16x32_f16(
              false, a[tm], false, b[tn], (short)0, acc[tm][tn], false, false);
    }
  };

  const int NK = K / BK;
  issue_async(0, 0);
  for (int kt = 0; kt < NK; kt++) {
    const int buf = kt & 1;
    wait_async_all();        // this wave's outstanding async copies done
    __syncthreads();         // all waves' copies visible; prior readers done
    if (kt + 1 < NK) issue_async(buf ^ 1, (kt + 1) * BK);  // overlap next tile
    compute(buf);
  }

  // Epilogue: C/D layout — VGPR i: lanes 0-15 -> row i, col lr; lanes 16-31 -> row 8+i
#pragma unroll
  for (int tm = 0; tm < 2; tm++) {
#pragma unroll
    for (int tn = 0; tn < 4; tn++) {
      const int n  = n0 + wn + tn * 16 + lr;
      const float bb = bias[n];
      const int mrow = m0 + wm + tm * 16 + lh * 8;
      float* po = out + (size_t)mrow * N + n;
#pragma unroll
      for (int i = 0; i < 8; i++)
        po[(size_t)i * N] = acc[tm][tn][i] + bb;
    }
  }
}

// ------------------------------------------------------------------
extern "C" void kernel_launch(void* const* d_in, const int* in_sizes, int n_in,
                              void* d_out, int out_size, void* d_ws, size_t ws_size,
                              hipStream_t stream) {
  const float* x  = (const float*)d_in[0];
  const int*   qw = (const int*)d_in[1];
  const float* sc = (const float*)d_in[2];
  const float* bs = (const float*)d_in[3];

  const int out_f = in_sizes[3];                       // 4096
  const int in_f  = in_sizes[1] / out_f;               // 4096
  const int ng    = in_sizes[2] / out_f;               // 32
  const int M     = (int)((long long)in_sizes[0] / in_f);  // 8192

  char* ws = (char*)d_ws;
  _Float16* xh = (_Float16*)ws;                                        // 64 MiB
  _Float16* wh = (_Float16*)(ws + (size_t)M * in_f * sizeof(_Float16)); // 32 MiB

  const size_t nx8 = (size_t)M * in_f / 8;
  cvt_x_f16_kernel<<<dim3((unsigned)((nx8 + 255) / 256)), 256, 0, stream>>>(x, xh, nx8);

  const size_t nw8 = (size_t)out_f * in_f / 8;
  dequant_w_f16_kernel<<<dim3((unsigned)((nw8 + 255) / 256)), 256, 0, stream>>>(
      qw, sc, wh, in_f, ng, nw8);

  const size_t lds_bytes = 4 * (size_t)TILE_ELEMS * sizeof(_Float16);  // 136 KiB
  dim3 grid(out_f / BN, M / BM);   // 32 x 64
  gptq_wmma_gemm_kernel<<<grid, THREADS, lds_bytes, stream>>>(
      xh, wh, bs, (float*)d_out, M, out_f, in_f);
}
